// SwarmNet_47510928228462
// MI455X (gfx1250) — compile-verified, matrix operand order checked by reading, not run on hardware
//
#include <hip/hip_runtime.h>
#include <hip/hip_bf16.h>

// Problem constants (reference: N=128 agents, T=128, D=8, H=32, T'=122)
#define NA   128
#define TT   128
#define DD   8
#define HH   32
#define TP   122
#define MM   (TP * NA)        // 15616 rows in the flattened [T',N] dim
#define MT   (MM / 16)        // 976 wave tiles
#define WPB  8                // waves per block for WMMA kernels (976 = 122*8)

typedef __attribute__((ext_vector_type(16))) _Float16 v16h;
typedef __attribute__((ext_vector_type(8)))  float    v8f;

// ---------------- WMMA fragment helpers (CDNA5 wave32 layouts) ----------------

// K index held by (lane, slot i) for 16-bit A(16x32) and B(32x16) fragments:
// lanes 0-15 hold K in {0..7, 16..23}; lanes 16-31 hold K in {8..15, 24..31}.
__device__ __forceinline__ int kmap(int lane, int i) {
  int kb = (lane >> 4) * 8;
  return (i < 8) ? (kb + i) : (16 + kb + (i - 8));
}

// Load A fragment: 16 consecutive rows (m0..m0+15) of a row-major f16 [M,32] buffer.
__device__ __forceinline__ v16h load_a_tile(const _Float16* __restrict__ h,
                                            int m0, int lane) {
  const _Float16* row = h + (size_t)(m0 + (lane & 15)) * HH;
  int kb = (lane >> 4) * 8;
  v16h a;
#pragma unroll
  for (int i = 0; i < 8; ++i) {
    a[i]     = row[kb + i];
    a[8 + i] = row[16 + kb + i];
  }
  return a;
}

// Load B fragment from an f32 weight matrix W (row-major, leading dim ldw):
// B[k][n] = W[(ncol0 + n) * ldw + koff + k]  (i.e. out = A @ W[ncol0:ncol0+16, koff:koff+32]^T)
__device__ __forceinline__ v16h load_b_w(const float* __restrict__ W, int ldw,
                                         int ncol0, int koff, int lane) {
  int n = ncol0 + (lane & 15);
  v16h b;
#pragma unroll
  for (int i = 0; i < 16; ++i)
    b[i] = (_Float16)W[n * ldw + koff + kmap(lane, i)];
  return b;
}

__device__ __forceinline__ v8f wmma_f16(v16h a, v16h b, v8f c) {
  // (neg_a, A, neg_b, B, c_mod, C, reuse_a, reuse_b)
  return __builtin_amdgcn_wmma_f32_16x16x32_f16(false, a, false, b, (short)0, c,
                                                false, false);
}

// ---------------- K1: conv stack (per-agent, LDS staged) ----------------
__global__ void k_conv(const float* __restrict__ x,
                       const float* __restrict__ w1, const float* __restrict__ b1,
                       const float* __restrict__ w2, const float* __restrict__ b2,
                       const float* __restrict__ w3, const float* __restrict__ b3,
                       _Float16* __restrict__ hF) {
  __shared__ float xS[TT * DD];        // 4 KB
  __shared__ float c1S[126 * 16];      // ~8 KB
  __shared__ float c2S[124 * 32];      // ~15.5 KB
  const int agent = blockIdx.x;
  const float* xp = x + (size_t)agent * TT * DD;

  for (int i = threadIdx.x; i < TT * DD; i += blockDim.x) xS[i] = xp[i];
  __syncthreads();

  // Conv1d(8->16, k=3) + ReLU
  for (int idx = threadIdx.x; idx < 126 * 16; idx += blockDim.x) {
    int t = idx >> 4, o = idx & 15;
    float acc = b1[o];
#pragma unroll
    for (int k = 0; k < 3; ++k)
#pragma unroll
      for (int d = 0; d < 8; ++d)
        acc += xS[(t + k) * 8 + d] * w1[o * 24 + d * 3 + k];
    c1S[idx] = fmaxf(acc, 0.f);
  }
  __syncthreads();

  // Conv1d(16->32, k=3) + ReLU
  for (int idx = threadIdx.x; idx < 124 * 32; idx += blockDim.x) {
    int t = idx >> 5, o = idx & 31;
    float acc = b2[o];
#pragma unroll
    for (int k = 0; k < 3; ++k)
#pragma unroll
      for (int c = 0; c < 16; ++c)
        acc += c1S[(t + k) * 16 + c] * w2[o * 48 + c * 3 + k];
    c2S[idx] = fmaxf(acc, 0.f);
  }
  __syncthreads();

  // Conv1d(32->32, k=3) + ReLU -> h[t, agent, c] in f16
  for (int idx = threadIdx.x; idx < TP * 32; idx += blockDim.x) {
    int t = idx >> 5, o = idx & 31;
    float acc = b3[o];
#pragma unroll
    for (int k = 0; k < 3; ++k)
#pragma unroll
      for (int c = 0; c < 32; ++c)
        acc += c2S[(t + k) * 32 + c] * w3[o * 96 + c * 3 + k];
    hF[((size_t)t * NA + agent) * HH + o] = (_Float16)fmaxf(acc, 0.f);
  }
}

// ---------------- K2: aj = h @ We[:,:32]^T ; ai = h @ We[:,32:]^T + be ----------------
__global__ void k_proj(const _Float16* __restrict__ hF,
                       const float* __restrict__ We, const float* __restrict__ be,
                       float* __restrict__ aj, float* __restrict__ ai) {
  const int lane = threadIdx.x & 31;
  const int tile = blockIdx.x * WPB + (threadIdx.x >> 5);
  const int m0 = tile * 16;

  v16h a   = load_a_tile(hF, m0, lane);
  v16h bj0 = load_b_w(We, 2 * HH, 0, 0, lane);
  v16h bj1 = load_b_w(We, 2 * HH, 16, 0, lane);
  v16h bi0 = load_b_w(We, 2 * HH, 0, HH, lane);
  v16h bi1 = load_b_w(We, 2 * HH, 16, HH, lane);

  v8f z = {};
  v8f cj0 = wmma_f16(a, bj0, z);
  v8f cj1 = wmma_f16(a, bj1, z);
  v8f ci0 = wmma_f16(a, bi0, z);
  v8f ci1 = wmma_f16(a, bi1, z);

  const int n = lane & 15, mh = 8 * (lane >> 4);
  const float be0 = be[n], be1 = be[n + 16];
#pragma unroll
  for (int r = 0; r < 8; ++r) {
    size_t row = (size_t)(m0 + r + mh) * HH;
    aj[row + n]      = cj0[r];
    aj[row + 16 + n] = cj1[r];
    ai[row + n]      = ci0[r] + be0;
    ai[row + 16 + n] = ci1[r] + be1;
  }
}

// ---------------- K3: edge aggregation (ReLU inside reduction -> VALU + LDS) ----------------
// agg0[t,i,k] = sum_j relu(aj[t,j,k] + ai[t,i,k]) - relu(aj[t,i,k] + ai[t,i,k])
__global__ void k_edge(const float* __restrict__ aj, const float* __restrict__ ai,
                       _Float16* __restrict__ agg0) {
  __shared__ float ajS[NA * HH];   // 16 KB
  __shared__ float aiS[NA * HH];   // 16 KB
  const int t = blockIdx.x;
  const float* ajp = aj + (size_t)t * NA * HH;
  const float* aip = ai + (size_t)t * NA * HH;
  for (int idx = threadIdx.x; idx < NA * HH; idx += blockDim.x) {
    ajS[idx] = ajp[idx];
    aiS[idx] = aip[idx];
  }
  __syncthreads();

  const int k = threadIdx.x & 31;      // channel = lane -> conflict-free LDS
  const int g = threadIdx.x >> 5;      // wave group: 8 groups x 16 agents
  float s[16], acc[16];
#pragma unroll
  for (int q = 0; q < 16; ++q) {
    s[q] = aiS[(g * 16 + q) * HH + k];
    acc[q] = 0.f;
  }
  for (int j = 0; j < NA; ++j) {
    float av = ajS[j * HH + k];
#pragma unroll
    for (int q = 0; q < 16; ++q) acc[q] += fmaxf(av + s[q], 0.f);
  }
#pragma unroll
  for (int q = 0; q < 16; ++q) {
    int i = g * 16 + q;
    float diag = fmaxf(ajS[i * HH + k] + s[q], 0.f);
    agg0[((size_t)t * NA + i) * HH + k] = (_Float16)(acc[q] - diag);
  }
}

// ---------------- K4: agg1 = relu(agg0 @ Wa^T + ba) ----------------
__global__ void k_agg(const _Float16* __restrict__ agg0,
                      const float* __restrict__ Wa, const float* __restrict__ ba,
                      _Float16* __restrict__ agg1) {
  const int lane = threadIdx.x & 31;
  const int tile = blockIdx.x * WPB + (threadIdx.x >> 5);
  const int m0 = tile * 16;

  v16h a  = load_a_tile(agg0, m0, lane);
  v16h b0 = load_b_w(Wa, HH, 0, 0, lane);
  v16h b1 = load_b_w(Wa, HH, 16, 0, lane);
  v8f z = {};
  v8f c0 = wmma_f16(a, b0, z);
  v8f c1 = wmma_f16(a, b1, z);

  const int n = lane & 15, mh = 8 * (lane >> 4);
  const float q0 = ba[n], q1 = ba[n + 16];
#pragma unroll
  for (int r = 0; r < 8; ++r) {
    size_t row = (size_t)(m0 + r + mh) * HH;
    agg1[row + n]      = (_Float16)fmaxf(c0[r] + q0, 0.f);
    agg1[row + 16 + n] = (_Float16)fmaxf(c1[r] + q1, 0.f);
  }
}

// ---------------- K5: upd = relu(h @ Wu[:,:32]^T + agg1 @ Wu[:,32:]^T + bu) ----------------
__global__ void k_upd(const _Float16* __restrict__ hF, const _Float16* __restrict__ agg1,
                      const float* __restrict__ Wu, const float* __restrict__ bu,
                      _Float16* __restrict__ upd) {
  const int lane = threadIdx.x & 31;
  const int tile = blockIdx.x * WPB + (threadIdx.x >> 5);
  const int m0 = tile * 16;

  v16h a1 = load_a_tile(hF, m0, lane);
  v16h a2 = load_a_tile(agg1, m0, lane);
  v16h b10 = load_b_w(Wu, 2 * HH, 0, 0, lane);
  v16h b11 = load_b_w(Wu, 2 * HH, 16, 0, lane);
  v16h b20 = load_b_w(Wu, 2 * HH, 0, HH, lane);
  v16h b21 = load_b_w(Wu, 2 * HH, 16, HH, lane);

  v8f z = {};
  v8f c0 = wmma_f16(a2, b20, wmma_f16(a1, b10, z));   // chained accumulation
  v8f c1 = wmma_f16(a2, b21, wmma_f16(a1, b11, z));

  const int n = lane & 15, mh = 8 * (lane >> 4);
  const float q0 = bu[n], q1 = bu[n + 16];
#pragma unroll
  for (int r = 0; r < 8; ++r) {
    size_t row = (size_t)(m0 + r + mh) * HH;
    upd[row + n]      = (_Float16)fmaxf(c0[r] + q0, 0.f);
    upd[row + 16 + n] = (_Float16)fmaxf(c1[r] + q1, 0.f);
  }
}

// ---------------- K6: dec = upd @ Wd^T + bd, residual with x[:,6+t,:], write out ----------------
__global__ void k_dec(const _Float16* __restrict__ upd,
                      const float* __restrict__ Wd, const float* __restrict__ bd,
                      const float* __restrict__ x, float* __restrict__ out) {
  const int lane = threadIdx.x & 31;
  const int tile = blockIdx.x * WPB + (threadIdx.x >> 5);
  const int m0 = tile * 16;
  const int n = lane & 15;

  v16h a = load_a_tile(upd, m0, lane);
  v16h b;
#pragma unroll
  for (int i = 0; i < 16; ++i)
    b[i] = (n < DD) ? (_Float16)Wd[n * HH + kmap(lane, i)] : (_Float16)0.f;

  v8f z = {};
  v8f c = wmma_f16(a, b, z);           // EXEC all-ones here; divergence only below

  if (n < DD) {
    const float bb = bd[n];
    const int mh = 8 * (lane >> 4);
#pragma unroll
    for (int r = 0; r < 8; ++r) {
      int m = m0 + r + mh;
      int t = m >> 7;                  // m = t*128 + agent
      int ag = m & 127;
      float res = x[(size_t)ag * TT * DD + (size_t)(t + 6) * DD + n];
      out[(size_t)ag * TP * DD + (size_t)t * DD + n] = c[r] + bb + res;
    }
  }
}

// ---------------- launch ----------------
extern "C" void kernel_launch(void* const* d_in, const int* in_sizes, int n_in,
                              void* d_out, int out_size, void* d_ws, size_t ws_size,
                              hipStream_t stream) {
  const float* x  = (const float*)d_in[0];
  const float* w1 = (const float*)d_in[1];
  const float* b1 = (const float*)d_in[2];
  const float* w2 = (const float*)d_in[3];
  const float* b2 = (const float*)d_in[4];
  const float* w3 = (const float*)d_in[5];
  const float* b3 = (const float*)d_in[6];
  const float* We = (const float*)d_in[7];
  const float* be = (const float*)d_in[8];
  const float* Wa = (const float*)d_in[9];
  const float* ba = (const float*)d_in[10];
  const float* Wu = (const float*)d_in[11];
  const float* bu = (const float*)d_in[12];
  const float* Wd = (const float*)d_in[13];
  const float* bd = (const float*)d_in[14];
  float* out = (float*)d_out;

  // Workspace carve-up (all MB-aligned; total 8 MB)
  char* ws = (char*)d_ws;
  _Float16* hF   = (_Float16*)(ws + 0);                    // [15616,32] f16
  float*    aj   = (float*)   (ws + (1u << 20));           // [15616,32] f32
  float*    ai   = (float*)   (ws + (3u << 20));           // [15616,32] f32
  _Float16* agg0 = (_Float16*)(ws + (5u << 20));           // [15616,32] f16
  _Float16* agg1 = (_Float16*)(ws + (6u << 20));           // [15616,32] f16
  _Float16* updb = (_Float16*)(ws + (7u << 20));           // [15616,32] f16

  k_conv<<<NA, 128, 0, stream>>>(x, w1, b1, w2, b2, w3, b3, hF);
  k_proj<<<MT / WPB, 32 * WPB, 0, stream>>>(hF, We, be, aj, ai);
  k_edge<<<TP, 256, 0, stream>>>(aj, ai, agg0);
  k_agg <<<MT / WPB, 32 * WPB, 0, stream>>>(agg0, Wa, ba, agg1);
  k_upd <<<MT / WPB, 32 * WPB, 0, stream>>>(hF, agg1, Wu, bu, updb);
  k_dec <<<MT / WPB, 32 * WPB, 0, stream>>>(updb, Wd, bd, x, out);
}
